// MetaLayer_17910013624370
// MI455X (gfx1250) — matrix operand order repaired and checked by reading, not compiled, not run on hardware
//
#include <hip/hip_runtime.h>

#define DN 128
#define NE 600000
#define NN 50000

typedef __attribute__((ext_vector_type(16))) __bf16 bf16x16;
typedef __attribute__((ext_vector_type(8)))  __bf16 bf16x8;
typedef __attribute__((ext_vector_type(8)))  float  f32x8;

__device__ __forceinline__ f32x8 wmma_bf16(bf16x16 a, bf16x16 b, f32x8 c) {
  // D = A(16x32 bf16) * B(32x16 bf16) + C(16x16 f32)
  return __builtin_amdgcn_wmma_f32_16x16x32_bf16(false, a, false, b, (short)0, c,
                                                 false, false);
}

__device__ __forceinline__ f32x8 zero8() {
  f32x8 z;
#pragma unroll
  for (int j = 0; j < 8; ++j) z[j] = 0.0f;
  return z;
}

// Pack A fragment from two contiguous 8-float chunks (global f32), scaled.
// A layout used here, contiguous k (16-bit 16x32): lane m=lane%16, half=lane/16:
//   elements 0..7  <- K = kb + 16*half + j
//   elements 8..15 <- K = kb + 16*half + 8 + j
__device__ __forceinline__ bf16x16 packA_f32(const float* p0, const float* p1, float s) {
  float4 a = *(const float4*)p0;
  float4 b = *(const float4*)(p0 + 4);
  float4 c = *(const float4*)p1;
  float4 d = *(const float4*)(p1 + 4);
  bf16x16 r;
  r[0]  = (__bf16)(a.x * s); r[1]  = (__bf16)(a.y * s);
  r[2]  = (__bf16)(a.z * s); r[3]  = (__bf16)(a.w * s);
  r[4]  = (__bf16)(b.x * s); r[5]  = (__bf16)(b.y * s);
  r[6]  = (__bf16)(b.z * s); r[7]  = (__bf16)(b.w * s);
  r[8]  = (__bf16)(c.x * s); r[9]  = (__bf16)(c.y * s);
  r[10] = (__bf16)(c.z * s); r[11] = (__bf16)(c.w * s);
  r[12] = (__bf16)(d.x * s); r[13] = (__bf16)(d.y * s);
  r[14] = (__bf16)(d.z * s); r[15] = (__bf16)(d.w * s);
  return r;
}

__device__ __forceinline__ bf16x16 packA_lds(const __bf16* p0, const __bf16* p1) {
  bf16x8 a = *(const bf16x8*)p0;
  bf16x8 b = *(const bf16x8*)p1;
  bf16x16 r;
#pragma unroll
  for (int j = 0; j < 8; ++j) { r[j] = a[j]; r[8 + j] = b[j]; }
  return r;
}

// B fragment: weights stored column-major bf16 (col n: K contiguous).
// lane n=lane%16, half=lane/16; element j <- W[kb + 16*half + j][n]
__device__ __forceinline__ bf16x16 loadB(const __bf16* p) {
  return *(const bf16x16*)p;
}

// ---------------- init: zero agg/cnt, repack weights fp32 row-major -> bf16 col-major
__global__ void init_kernel(const float* __restrict__ We1, const float* __restrict__ We2,
                            const float* __restrict__ Wn1, const float* __restrict__ Wn2,
                            float* __restrict__ agg, float* __restrict__ cnt,
                            __bf16* __restrict__ w1, __bf16* __restrict__ w2,
                            __bf16* __restrict__ wn1, __bf16* __restrict__ wn2) {
  long t = (long)blockIdx.x * blockDim.x + threadIdx.x;
  long stride = (long)gridDim.x * blockDim.x;
  for (long i = t; i < (long)NN * DN; i += stride) agg[i] = 0.0f;
  for (long i = t; i < NN; i += stride) cnt[i] = 0.0f;
  for (long i = t; i < 384 * DN; i += stride) {
    long k = i / DN, n = i % DN;
    w1[n * 384 + k] = (__bf16)We1[i];
  }
  for (long i = t; i < 128 * DN; i += stride) {
    long k = i / DN, n = i % DN;
    w2[n * 128 + k] = (__bf16)We2[i];
  }
  for (long i = t; i < 256 * DN; i += stride) {
    long k = i / DN, n = i % DN;
    wn1[n * 256 + k] = (__bf16)Wn1[i];
  }
  for (long i = t; i < 128 * DN; i += stride) {
    long k = i / DN, n = i % DN;
    wn2[n * 128 + k] = (__bf16)Wn2[i];
  }
}

// ---------------- edge MLP: e = relu([x_s|x_d|ea] @ W1 + b1) @ W2 + b2
// plus atomic scatter of e into agg[dest], count into cnt[dest]
__global__ __launch_bounds__(256) void edge_kernel(
    const float* __restrict__ x, const int* __restrict__ ei, const float* __restrict__ ea,
    const __bf16* __restrict__ W1, const float* __restrict__ b1,
    const __bf16* __restrict__ W2, const float* __restrict__ b2,
    float* __restrict__ e_out, float* __restrict__ agg, float* __restrict__ cnt) {
  __shared__ __align__(32) __bf16 hbuf[8][32 * DN];  // 64 KB / block
  const int wv = threadIdx.x >> 5;
  const int lane = threadIdx.x & 31;
  const int m = lane & 15;
  const int half = lane >> 4;
  const long wid = (long)blockIdx.x * 8 + wv;
  const long e0 = wid * 32;
  if (e0 >= NE) return;

  int srcm[2], dstm[2];
#pragma unroll
  for (int mt = 0; mt < 2; ++mt) {
    long er = e0 + mt * 16 + m;
    srcm[mt] = ei[er];
    dstm[mt] = ei[NE + er];
  }

  // ---- GEMM1: [32 x 384] @ [384 x 128], software-pipelined one k-step ahead
  f32x8 acc[2][8];
#pragma unroll
  for (int mt = 0; mt < 2; ++mt)
#pragma unroll
    for (int nt = 0; nt < 8; ++nt) acc[mt][nt] = zero8();

  auto loadA1 = [&](int kb, bf16x16 a[2]) {
#pragma unroll
    for (int mt = 0; mt < 2; ++mt) {
      const float* row;
      if (kb < 128)      row = x  + (long)srcm[mt] * DN + kb;
      else if (kb < 256) row = x  + (long)dstm[mt] * DN + (kb - 128);
      else               row = ea + (e0 + mt * 16 + m) * DN + (kb - 256);
      const float* p0 = row + 16 * half;
      a[mt] = packA_f32(p0, p0 + 8, 1.0f);
    }
  };
  auto loadB1 = [&](int kb, bf16x16 bfr[8]) {
#pragma unroll
    for (int nt = 0; nt < 8; ++nt)
      bfr[nt] = loadB(W1 + (long)(nt * 16 + m) * 384 + kb + 16 * half);
  };

  {
    bf16x16 acur[2], anxt[2], bcur[8], bnxt[8];
    loadB1(0, bcur);
    loadA1(0, acur);
#pragma unroll
    for (int ks = 0; ks < 12; ++ks) {
      if (ks < 11) {
        loadB1((ks + 1) * 32, bnxt);
        loadA1((ks + 1) * 32, anxt);
      }
#pragma unroll
      for (int nt = 0; nt < 8; ++nt) {
        acc[0][nt] = wmma_bf16(acur[0], bcur[nt], acc[0][nt]);
        acc[1][nt] = wmma_bf16(acur[1], bcur[nt], acc[1][nt]);
      }
#pragma unroll
      for (int nt = 0; nt < 8; ++nt) bcur[nt] = bnxt[nt];
      acur[0] = anxt[0];
      acur[1] = anxt[1];
    }
  }

  // ---- bias + relu -> LDS (C layout: elem i = row i+8*half, col = nt*16+m)
  __bf16* hb = &hbuf[wv][0];
#pragma unroll
  for (int mt = 0; mt < 2; ++mt)
#pragma unroll
    for (int nt = 0; nt < 8; ++nt) {
      const float bias = b1[nt * 16 + m];
#pragma unroll
      for (int i = 0; i < 8; ++i) {
        float v = acc[mt][nt][i] + bias;
        v = fmaxf(v, 0.0f);
        hb[(mt * 16 + i + 8 * half) * DN + nt * 16 + m] = (__bf16)v;
      }
    }
  // same-wave LDS ops are in-order; compiler inserts s_wait_dscnt for the reads

  // ---- GEMM2: [32 x 128] @ [128 x 128], pipelined
  f32x8 acc2[2][8];
#pragma unroll
  for (int mt = 0; mt < 2; ++mt)
#pragma unroll
    for (int nt = 0; nt < 8; ++nt) acc2[mt][nt] = zero8();

  auto loadA2 = [&](int kb, bf16x16 a[2]) {
#pragma unroll
    for (int mt = 0; mt < 2; ++mt) {
      const __bf16* p0 = hb + (mt * 16 + m) * DN + kb + 16 * half;
      a[mt] = packA_lds(p0, p0 + 8);
    }
  };
  auto loadB2 = [&](int kb, bf16x16 bfr[8]) {
#pragma unroll
    for (int nt = 0; nt < 8; ++nt)
      bfr[nt] = loadB(W2 + (long)(nt * 16 + m) * 128 + kb + 16 * half);
  };

  {
    bf16x16 acur[2], anxt[2], bcur[8], bnxt[8];
    loadB2(0, bcur);
    loadA2(0, acur);
#pragma unroll
    for (int ks = 0; ks < 4; ++ks) {
      if (ks < 3) {
        loadB2((ks + 1) * 32, bnxt);
        loadA2((ks + 1) * 32, anxt);
      }
#pragma unroll
      for (int nt = 0; nt < 8; ++nt) {
        acc2[0][nt] = wmma_bf16(acur[0], bcur[nt], acc2[0][nt]);
        acc2[1][nt] = wmma_bf16(acur[1], bcur[nt], acc2[1][nt]);
      }
#pragma unroll
      for (int nt = 0; nt < 8; ++nt) bcur[nt] = bnxt[nt];
      acur[0] = anxt[0];
      acur[1] = anxt[1];
    }
  }

  // ---- epilogue: bias, store e, atomic scatter into agg/cnt
#pragma unroll
  for (int mt = 0; mt < 2; ++mt) {
    int dvr[8];
#pragma unroll
    for (int i = 0; i < 8; ++i) dvr[i] = __shfl(dstm[mt], i + 8 * half, 32);
#pragma unroll
    for (int nt = 0; nt < 8; ++nt) {
      const float bias = b2[nt * 16 + m];
#pragma unroll
      for (int i = 0; i < 8; ++i) {
        float v = acc2[mt][nt][i] + bias;
        long er = e0 + mt * 16 + i + 8 * half;
        e_out[er * DN + nt * 16 + m] = v;
        atomicAdd(&agg[(long)dvr[i] * DN + nt * 16 + m], v);
      }
    }
    if (half == 0) atomicAdd(&cnt[dstm[mt]], 1.0f);
  }
}

// ---------------- node MLP: x_out = relu([x | agg/max(cnt,1)] @ Wn1 + bn1) @ Wn2 + bn2
__global__ __launch_bounds__(256) void node_kernel(
    const float* __restrict__ x, const float* __restrict__ agg, const float* __restrict__ cnt,
    const __bf16* __restrict__ W1, const float* __restrict__ b1,
    const __bf16* __restrict__ W2, const float* __restrict__ b2,
    float* __restrict__ xout) {
  __shared__ __align__(32) __bf16 hbuf[8][32 * DN];
  const int wv = threadIdx.x >> 5;
  const int lane = threadIdx.x & 31;
  const int m = lane & 15;
  const int half = lane >> 4;
  const long wid = (long)blockIdx.x * 8 + wv;
  const long n0 = wid * 32;
  if (n0 >= NN) return;

  int nodec[2];
  float sc0[2], sc1[2];
#pragma unroll
  for (int mt = 0; mt < 2; ++mt) {
    int node = (int)(n0 + mt * 16 + m);
    bool valid = node < NN;
    nodec[mt] = valid ? node : 0;
    sc0[mt] = valid ? 1.0f : 0.0f;
    float c = cnt[nodec[mt]];
    sc1[mt] = valid ? (1.0f / fmaxf(c, 1.0f)) : 0.0f;
  }

  // ---- GEMM1: [32 x 256] @ [256 x 128], pipelined
  f32x8 acc[2][8];
#pragma unroll
  for (int mt = 0; mt < 2; ++mt)
#pragma unroll
    for (int nt = 0; nt < 8; ++nt) acc[mt][nt] = zero8();

  auto loadA1 = [&](int kb, bf16x16 a[2]) {
#pragma unroll
    for (int mt = 0; mt < 2; ++mt) {
      const float* row;
      float s;
      if (kb < 128) { row = x   + (long)nodec[mt] * DN + kb;         s = sc0[mt]; }
      else          { row = agg + (long)nodec[mt] * DN + (kb - 128); s = sc1[mt]; }
      const float* p0 = row + 16 * half;
      a[mt] = packA_f32(p0, p0 + 8, s);
    }
  };
  auto loadB1 = [&](int kb, bf16x16 bfr[8]) {
#pragma unroll
    for (int nt = 0; nt < 8; ++nt)
      bfr[nt] = loadB(W1 + (long)(nt * 16 + m) * 256 + kb + 16 * half);
  };

  {
    bf16x16 acur[2], anxt[2], bcur[8], bnxt[8];
    loadB1(0, bcur);
    loadA1(0, acur);
#pragma unroll
    for (int ks = 0; ks < 8; ++ks) {
      if (ks < 7) {
        loadB1((ks + 1) * 32, bnxt);
        loadA1((ks + 1) * 32, anxt);
      }
#pragma unroll
      for (int nt = 0; nt < 8; ++nt) {
        acc[0][nt] = wmma_bf16(acur[0], bcur[nt], acc[0][nt]);
        acc[1][nt] = wmma_bf16(acur[1], bcur[nt], acc[1][nt]);
      }
#pragma unroll
      for (int nt = 0; nt < 8; ++nt) bcur[nt] = bnxt[nt];
      acur[0] = anxt[0];
      acur[1] = anxt[1];
    }
  }

  __bf16* hb = &hbuf[wv][0];
#pragma unroll
  for (int mt = 0; mt < 2; ++mt)
#pragma unroll
    for (int nt = 0; nt < 8; ++nt) {
      const float bias = b1[nt * 16 + m];
#pragma unroll
      for (int i = 0; i < 8; ++i) {
        float v = acc[mt][nt][i] + bias;
        v = fmaxf(v, 0.0f);
        hb[(mt * 16 + i + 8 * half) * DN + nt * 16 + m] = (__bf16)v;
      }
    }

  // ---- GEMM2: [32 x 128] @ [128 x 128], pipelined
  f32x8 acc2[2][8];
#pragma unroll
  for (int mt = 0; mt < 2; ++mt)
#pragma unroll
    for (int nt = 0; nt < 8; ++nt) acc2[mt][nt] = zero8();

  auto loadA2 = [&](int kb, bf16x16 a[2]) {
#pragma unroll
    for (int mt = 0; mt < 2; ++mt) {
      const __bf16* p0 = hb + (mt * 16 + m) * DN + kb + 16 * half;
      a[mt] = packA_lds(p0, p0 + 8);
    }
  };
  auto loadB2 = [&](int kb, bf16x16 bfr[8]) {
#pragma unroll
    for (int nt = 0; nt < 8; ++nt)
      bfr[nt] = loadB(W2 + (long)(nt * 16 + m) * 128 + kb + 16 * half);
  };

  {
    bf16x16 acur[2], anxt[2], bcur[8], bnxt[8];
    loadB2(0, bcur);
    loadA2(0, acur);
#pragma unroll
    for (int ks = 0; ks < 4; ++ks) {
      if (ks < 3) {
        loadB2((ks + 1) * 32, bnxt);
        loadA2((ks + 1) * 32, anxt);
      }
#pragma unroll
      for (int nt = 0; nt < 8; ++nt) {
        acc2[0][nt] = wmma_bf16(acur[0], bcur[nt], acc2[0][nt]);
        acc2[1][nt] = wmma_bf16(acur[1], bcur[nt], acc2[1][nt]);
      }
#pragma unroll
      for (int nt = 0; nt < 8; ++nt) bcur[nt] = bnxt[nt];
      acur[0] = anxt[0];
      acur[1] = anxt[1];
    }
  }

#pragma unroll
  for (int mt = 0; mt < 2; ++mt)
#pragma unroll
    for (int nt = 0; nt < 8; ++nt) {
      const float bias = b2[nt * 16 + m];
#pragma unroll
      for (int i = 0; i < 8; ++i) {
        long nr = n0 + mt * 16 + i + 8 * half;
        if (nr < NN) {
          xout[nr * DN + nt * 16 + m] = acc2[mt][nt][i] + bias;
        }
      }
    }
}

extern "C" void kernel_launch(void* const* d_in, const int* in_sizes, int n_in,
                              void* d_out, int out_size, void* d_ws, size_t ws_size,
                              hipStream_t stream) {
  const float* x   = (const float*)d_in[0];
  const int*   ei  = (const int*)d_in[1];
  const float* ea  = (const float*)d_in[2];
  const float* We1 = (const float*)d_in[3];
  const float* be1 = (const float*)d_in[4];
  const float* We2 = (const float*)d_in[5];
  const float* be2 = (const float*)d_in[6];
  const float* Wn1 = (const float*)d_in[7];
  const float* bn1 = (const float*)d_in[8];
  const float* Wn2 = (const float*)d_in[9];
  const float* bn2 = (const float*)d_in[10];

  float* xout = (float*)d_out;
  float* eout = xout + (size_t)NN * DN;

  // workspace layout
  char* ws = (char*)d_ws;
  size_t off = 0;
  auto take = [&](size_t bytes) {
    size_t cur = off;
    off = (off + bytes + 255) & ~(size_t)255;
    return cur;
  };
  float* agg  = (float*)(ws + take((size_t)NN * DN * 4));
  float* cnt  = (float*)(ws + take((size_t)NN * 4));
  __bf16* w1  = (__bf16*)(ws + take((size_t)384 * DN * 2));
  __bf16* w2  = (__bf16*)(ws + take((size_t)128 * DN * 2));
  __bf16* wn1 = (__bf16*)(ws + take((size_t)256 * DN * 2));
  __bf16* wn2 = (__bf16*)(ws + take((size_t)128 * DN * 2));
  (void)ws_size; (void)in_sizes; (void)n_in; (void)out_size;

  init_kernel<<<1024, 256, 0, stream>>>(We1, We2, Wn1, Wn2, agg, cnt, w1, w2, wn1, wn2);

  // 600000 edges / 32 per wave = 18750 waves; 8 waves per block
  int edge_blocks = (18750 + 7) / 8;  // 2344
  edge_kernel<<<edge_blocks, 256, 0, stream>>>(x, ei, ea, w1, be1, w2, be2, eout, agg, cnt);

  // 50000 nodes / 32 per wave = 1563 waves; 8 waves per block
  int node_blocks = (1563 + 7) / 8;  // 196
  node_kernel<<<node_blocks, 256, 0, stream>>>(x, agg, cnt, wn1, bn1, wn2, bn2, xout);
}
